// ParticleNet_69707319214173
// MI455X (gfx1250) — matrix operand order repaired and checked by reading, not compiled, not run on hardware
//
#include <hip/hip_runtime.h>
#include <hip/hip_bf16.h>
#include <math.h>

// Problem constants (from reference): B=1024, P=128, F=16, K=4, NC=5
#define BN_  1024
#define PN_  128
#define FN_  16
#define KN_  4
#define NC_  5
#define NN_  (BN_*PN_)     // 131072 nodes
#define EN_  (NN_*KN_)     // 524288 edges
#define EPSF 1e-5f

typedef __attribute__((ext_vector_type(16))) _Float16 v16h;
typedef __attribute__((ext_vector_type(8)))  _Float16 v8h;
typedef __attribute__((ext_vector_type(8)))  float    v8f;

struct MlpLayerP { const float *lb, *bb, *bg, *bm, *bv; };   // bias + BN params
struct ConvP     { MlpLayerP L0, L1, L2;
                   const _Float16 *q0, *q1, *q2;             // packed f16 weights
                   const float *sc_w, *sc_b, *sc_bb, *sc_bg, *sc_bm, *sc_bv; };
struct HeadP     { MlpLayerP l1, l2;
                   const _Float16 *q1, *q2;
                   const float *out_b, *out_w; };

// ---------------- WMMA fragment helpers (contiguous k = 16*(lane/16) + s per lane, wave32) -------

// A fragment: 16x32 f16 row-major, stride lda halves. Per lane: two contiguous
// 16B runs -> 2x ds_load_b128.
__device__ __forceinline__ v16h load_a_frag(const _Float16* A, int lda) {
  const int lane = threadIdx.x & 31;
  const _Float16* p = A + (lane & 15)*lda + (lane >> 4)*16;
  v8h lo = *(const v8h*)(p);
  v8h hi = *(const v8h*)(p + 8);
  return __builtin_shufflevector(lo, hi, 0,1,2,3,4,5,6,7,8,9,10,11,12,13,14,15);
}

// B fragment for X^T (Gram): B[k][q] = X[q][k]; per lane 32B contiguous.
__device__ __forceinline__ v16h load_bT_frag(const _Float16* X, int ldx) {
  const int lane = threadIdx.x & 31;
  return *(const v16h*)(X + (lane & 15)*ldx + (lane >> 4)*16);
}

// ---------- Lin(+bias) -> ReLU -> BN layer on a 64-row LDS tile ------------
// 128 threads = 4 waves; wave w owns rows [16w,16w+16).
// Wq: weights pre-swizzled as [kstep][ntile][lane][16 halves] (32B/lane/frag),
// read directly from global (L2-resident, ~274KB total for the whole net).
template<int K, int COUT>
__device__ __forceinline__ void mlp_layer(const _Float16* __restrict__ Ain,
                                          const _Float16* __restrict__ Wq,
                                          const MlpLayerP& P,
                                          _Float16* __restrict__ Aout)
{
  constexpr int NT  = COUT / 16;
  constexpr int NKS = K / 32;
  const int tid  = threadIdx.x;
  const int wave = tid >> 5;
  const int lane = tid & 31;

  v8f zero;
#pragma unroll
  for (int r = 0; r < 8; ++r) zero[r] = 0.f;
  v8f acc[NT];
#pragma unroll
  for (int t = 0; t < NT; ++t) acc[t] = zero;

#pragma unroll
  for (int ks = 0; ks < NKS; ++ks) {
    v16h a = load_a_frag(Ain + wave*16*K + ks*32, K);
    const _Float16* wp = Wq + (size_t)ks*NT*512 + lane*16;
#pragma unroll
    for (int t = 0; t < NT; ++t) {
      v16h b = *(const v16h*)(wp + t*512);
      acc[t] = __builtin_amdgcn_wmma_f32_16x16x32_f16(false, a, false, b,
                                                      (short)0, acc[t], false, false);
    }
  }
  // epilogue: bias + ReLU + BN -> f16 (C/D layout: VGPR r, lane-half selects M)
  const int n  = lane & 15;
  const int mb = wave*16 + ((lane >> 4) * 8);
#pragma unroll
  for (int t = 0; t < NT; ++t) {
    const int ch = t*16 + n;
    const float scale = P.bg[ch] * rsqrtf(P.bv[ch] + EPSF);
    const float shift = P.bb[ch] - P.bm[ch] * scale;
    const float bias  = P.lb[ch];
#pragma unroll
    for (int r = 0; r < 8; ++r) {
      float y = acc[t][r] + bias;
      y = fmaxf(y, 0.f);
      y = y * scale + shift;
      Aout[(mb + r)*COUT + ch] = (_Float16)y;
    }
  }
  __syncthreads();
}

// ------------- weight pre-pack: f32 KxCOUT -> swizzled f16 fragments -------
template<int K, int COUT>
__global__ __launch_bounds__(256) void pack_w_kernel(const float* __restrict__ W,
                                                     _Float16* __restrict__ Wq)
{
  constexpr int NT = COUT / 16;
  const int j = blockIdx.x*256 + threadIdx.x;
  if (j >= K*COUT) return;
  const int ks   = j / (NT*512);
  const int rem  = j % (NT*512);
  const int t    = rem / 512;
  const int lane = (rem % 512) / 16;
  const int h    = rem % 16;
  const int k = ks*32 + (lane >> 4)*16 + h;
  const int n = t*16 + (lane & 15);
  Wq[j] = (_Float16)W[(size_t)k*COUT + n];
}

// ------------------------- group norm (per batch) --------------------------
__global__ __launch_bounds__(128) void gn_kernel(const float* __restrict__ x,
    const float* __restrict__ gb, const float* __restrict__ gg,
    const float* __restrict__ gms, _Float16* __restrict__ out)
{
  __shared__ float s[PN_][FN_];
  __shared__ float mean_s[FN_], var_s[FN_];
  const int b = blockIdx.x, p = threadIdx.x;
  const size_t base = ((size_t)b*PN_ + p)*FN_;
  float v[FN_];
  for (int f = 0; f < FN_; ++f) { v[f] = x[base + f]; s[p][f] = v[f]; }
  __syncthreads();
  for (int st = 64; st > 0; st >>= 1) {
    if (p < st) for (int f = 0; f < FN_; ++f) s[p][f] += s[p + st][f];
    __syncthreads();
  }
  if (p < FN_) mean_s[p] = s[0][p] * (1.f/PN_);
  __syncthreads();
  float c[FN_];
  for (int f = 0; f < FN_; ++f) { c[f] = v[f] - gms[f]*mean_s[f]; s[p][f] = c[f]*c[f]; }
  __syncthreads();
  for (int st = 64; st > 0; st >>= 1) {
    if (p < st) for (int f = 0; f < FN_; ++f) s[p][f] += s[p + st][f];
    __syncthreads();
  }
  if (p < FN_) var_s[p] = s[0][p] * (1.f/PN_);
  __syncthreads();
  for (int f = 0; f < FN_; ++f)
    out[base + f] = (_Float16)(gg[f]*c[f]*rsqrtf(var_s[f] + EPSF) + gb[f]);
}

// --------- EdgeConv MLP over 64 edges (16 nodes * K=4) per block -----------
template<int CIN, int CH>
__global__ __launch_bounds__(128) void edge_mlp_kernel(const _Float16* __restrict__ hin,
    const int* __restrict__ srcIdx, ConvP cp, _Float16* __restrict__ hout)
{
  constexpr int CCAT = 2*CIN;
  constexpr int W0   = (CCAT > CH) ? CCAT : CH;
  constexpr int CV   = CIN/8;                 // v8h chunks per row
  __shared__ alignas(32) _Float16 A0[64*W0];
  __shared__ alignas(32) _Float16 A1[64*CH];
  __shared__ int s_src[64];

  const int tid = threadIdx.x;
  const int nb  = blockIdx.x * 16;            // first node of this block
  __builtin_prefetch(cp.q0, 0, 3);
  if (tid < 64) s_src[tid] = srcIdx[(size_t)nb*KN_ + tid];
  __syncthreads();
  // build [xi, xj-xi] with packed-f16 vector ops
  for (int i = tid; i < 64*CV; i += 128) {
    const int r = i / CV, c8 = i % CV;
    const int node = nb + (r >> 2);
    const int s    = s_src[r];
    v8h xi = *(const v8h*)(hin + (size_t)node*CIN + c8*8);
    v8h xj = *(const v8h*)(hin + (size_t)s   *CIN + c8*8);
    *(v8h*)(A0 + r*CCAT + c8*8)       = xi;
    *(v8h*)(A0 + r*CCAT + CIN + c8*8) = xj - xi;
  }
  __syncthreads();
  mlp_layer<CCAT, CH>(A0, cp.q0, cp.L0, A1);
  mlp_layer<CH,   CH>(A1, cp.q1, cp.L1, A0);
  mlp_layer<CH,   CH>(A0, cp.q2, cp.L2, A1);
  // mean over K=4 neighbors + BN(Lin_sc(x)) shortcut
  for (int i = tid; i < 16*CH; i += 128) {
    const int ni = i / CH, c = i % CH;
    const int node = nb + ni;
    float m = 0.f;
#pragma unroll
    for (int k = 0; k < KN_; ++k) m += (float)A1[(ni*KN_ + k)*CH + c];
    m *= 0.25f;
    float a = cp.sc_b[c];
    for (int k = 0; k < CIN; ++k)
      a += (float)hin[(size_t)node*CIN + k] * cp.sc_w[(size_t)k*CH + c];
    const float scale = cp.sc_bg[c]*rsqrtf(cp.sc_bv[c] + EPSF);
    const float sc    = (a - cp.sc_bm[c])*scale + cp.sc_bb[c];
    hout[(size_t)node*CH + c] = (_Float16)(m + sc);
  }
}

// ------------- KNN per batch via WMMA Gram matrix (d2 = n2p+n2q-2G) --------
template<int C>
__global__ __launch_bounds__(128) void knn_kernel(const _Float16* __restrict__ h,
                                                  int* __restrict__ idx)
{
  __shared__ alignas(32) _Float16 X[PN_*C];
  __shared__ float G[PN_*16];
  __shared__ float n2[PN_];
  const int b = blockIdx.x, tid = threadIdx.x;
  const int wave = tid >> 5;
  __builtin_prefetch(h + (size_t)b*PN_*C, 0, 3);
  for (int i = tid; i < PN_*C/8; i += 128)
    ((v8h*)X)[i] = ((const v8h*)(h + (size_t)b*PN_*C))[i];
  __syncthreads();
  {
    float s = 0.f;
    for (int k = 0; k < C; ++k) { float v = (float)X[tid*C + k]; s += v*v; }
    n2[tid] = s;
  }
  __syncthreads();
  float bd0=3e38f, bd1=3e38f, bd2=3e38f, bd3=3e38f;
  int   bi0=0, bi1=0, bi2=0, bi3=0;
  for (int qb = 0; qb < 8; ++qb) {            // 16-column block of Gram
    for (int mt = wave; mt < 8; mt += 4) {    // 2 M-tiles per wave
      v8f acc;
#pragma unroll
      for (int r = 0; r < 8; ++r) acc[r] = 0.f;
#pragma unroll
      for (int k0 = 0; k0 < C; k0 += 32) {
        v16h a  = load_a_frag (X + mt*16*C + k0, C);
        v16h bf = load_bT_frag(X + qb*16*C + k0, C);
        acc = __builtin_amdgcn_wmma_f32_16x16x32_f16(false, a, false, bf,
                                                     (short)0, acc, false, false);
      }
      const int lane = tid & 31;
      const int n  = lane & 15;
      const int mb = mt*16 + ((lane >> 4) * 8);
#pragma unroll
      for (int r = 0; r < 8; ++r) G[(mb + r)*16 + n] = acc[r];
    }
    __syncthreads();
    for (int j = 0; j < 16; ++j) {            // top-4 smallest d2 (self excluded)
      const int q = qb*16 + j;
      if (q == tid) continue;
      const float d = n2[q] - 2.f*G[tid*16 + j];  // + n2[tid] is rank-invariant
      if (d < bd3) {
        bd3 = d; bi3 = q;
        if (bd3 < bd2) { float t=bd2; bd2=bd3; bd3=t; int ti=bi2; bi2=bi3; bi3=ti; }
        if (bd2 < bd1) { float t=bd1; bd1=bd2; bd2=t; int ti=bi1; bi1=bi2; bi2=ti; }
        if (bd1 < bd0) { float t=bd0; bd0=bd1; bd1=t; int ti=bi0; bi0=bi1; bi1=ti; }
      }
    }
    __syncthreads();
  }
  int* op = idx + ((size_t)b*PN_ + tid)*KN_;
  op[0] = b*PN_ + bi0; op[1] = b*PN_ + bi1;
  op[2] = b*PN_ + bi2; op[3] = b*PN_ + bi3;
}

// ------------------------- global mean pool --------------------------------
__global__ __launch_bounds__(128) void pool_kernel(const _Float16* __restrict__ h,
                                                   _Float16* __restrict__ gout)
{
  const int b = blockIdx.x, c = threadIdx.x;
  float s = 0.f;
  for (int p = 0; p < PN_; ++p) s += (float)h[((size_t)b*PN_ + p)*128 + c];
  gout[(size_t)b*128 + c] = (_Float16)(s * (1.f/PN_));
}

// ------------------------- head MLP + softmax ------------------------------
__global__ __launch_bounds__(128) void head_kernel(const _Float16* __restrict__ gin,
                                                   HeadP hp, float* __restrict__ out)
{
  __shared__ alignas(32) _Float16 A0[64*128];
  __shared__ alignas(32) _Float16 A1[64*64];
  __shared__ alignas(32) _Float16 A2[64*64];
  const int tid = threadIdx.x;
  const int rb  = blockIdx.x * 64;
  for (int i = tid; i < 64*128/8; i += 128)
    ((v8h*)A0)[i] = ((const v8h*)(gin + (size_t)rb*128))[i];
  __syncthreads();
  mlp_layer<128, 64>(A0, hp.q1, hp.l1, A1);
  mlp_layer<64,  64>(A1, hp.q2, hp.l2, A2);
  if (tid < 64) {
    float lg[NC_]; float mx = -3e38f;
    for (int j = 0; j < NC_; ++j) {
      float a = hp.out_b[j];
      for (int k = 0; k < 64; ++k) a += (float)A2[tid*64 + k] * hp.out_w[k*NC_ + j];
      lg[j] = a; mx = fmaxf(mx, a);
    }
    float den = 0.f;
    for (int j = 0; j < NC_; ++j) { lg[j] = __expf(lg[j] - mx); den += lg[j]; }
    const float inv = 1.f/den;
    for (int j = 0; j < NC_; ++j) out[(size_t)(rb + tid)*NC_ + j] = lg[j]*inv;
  }
}

// ------------------------------ launch -------------------------------------
// Input pytree flattening assumed (sorted dict keys per level after x, edge_index):
// 0:x 1:edge_index
// per conv (24 leaves): mlp{0,1,2}:{lin.b, lin.w, bn.b, bn.g, bn.m, bn.v},
//   sc_bn.{b,g,m,v}, sc_lin.{b,w}
// 2..25:conv1  26..49:conv2  50..73:conv3  74..76:gn.{b,g,ms}
// 77..90:head bn1.{b,g,m,v} bn2.{b,g,m,v} d1.{b,w} d2.{b,w} out.{b,w}
static inline MlpLayerP mk_layer(void* const* d, int base) {
  MlpLayerP p;
  p.lb = (const float*)d[base+0];
  p.bb = (const float*)d[base+2]; p.bg = (const float*)d[base+3];
  p.bm = (const float*)d[base+4]; p.bv = (const float*)d[base+5];
  return p;
}
static inline ConvP mk_conv(void* const* d, int base) {
  ConvP c;
  c.L0 = mk_layer(d, base+0); c.L1 = mk_layer(d, base+6); c.L2 = mk_layer(d, base+12);
  c.sc_bb = (const float*)d[base+18]; c.sc_bg = (const float*)d[base+19];
  c.sc_bm = (const float*)d[base+20]; c.sc_bv = (const float*)d[base+21];
  c.sc_b  = (const float*)d[base+22]; c.sc_w  = (const float*)d[base+23];
  return c;
}

// packed-weight offsets (in halves) inside the ws weight region
#define O10 0        // conv1 L0: 32x64
#define O11 2048     // conv1 L1: 64x64
#define O12 6144     // conv1 L2: 64x64
#define O20 10240    // conv2 L0: 128x128
#define O21 26624    // conv2 L1
#define O22 43008    // conv2 L2
#define O30 59392    // conv3 L0: 256x128
#define O31 92160    // conv3 L1
#define O32 108544   // conv3 L2
#define OH1 124928   // head d1: 128x64
#define OH2 133120   // head d2: 64x64
#define OWTOT 137216

extern "C" void kernel_launch(void* const* d_in, const int* in_sizes, int n_in,
                              void* d_out, int out_size, void* d_ws, size_t ws_size,
                              hipStream_t stream)
{
  if (n_in < 91) return;
  const float* x  = (const float*)d_in[0];
  const int*   ei = (const int*)d_in[1];          // [0:E)=src, [E:2E)=dst
  ConvP c1 = mk_conv(d_in, 2);
  ConvP c2 = mk_conv(d_in, 26);
  ConvP c3 = mk_conv(d_in, 50);
  const float* gn_b  = (const float*)d_in[74];
  const float* gn_g  = (const float*)d_in[75];
  const float* gn_ms = (const float*)d_in[76];
  HeadP hp;
  hp.l1.bb=(const float*)d_in[77]; hp.l1.bg=(const float*)d_in[78];
  hp.l1.bm=(const float*)d_in[79]; hp.l1.bv=(const float*)d_in[80];
  hp.l2.bb=(const float*)d_in[81]; hp.l2.bg=(const float*)d_in[82];
  hp.l2.bm=(const float*)d_in[83]; hp.l2.bv=(const float*)d_in[84];
  hp.l1.lb=(const float*)d_in[85];
  hp.l2.lb=(const float*)d_in[87];
  hp.out_b=(const float*)d_in[89]; hp.out_w=(const float*)d_in[90];

  // Workspace layout (f16 activations + packed weights)
  char* ws = (char*)d_ws;
  _Float16* x16 = (_Float16*)(ws + 0);                            //  4 MB  N*16
  _Float16* h1  = (_Float16*)(ws + (size_t)4*1024*1024);          // 16 MB  N*64
  _Float16* h2  = (_Float16*)(ws + (size_t)20*1024*1024);         // 32 MB  N*128
  _Float16* h3  = (_Float16*)(ws + (size_t)52*1024*1024);         // 32 MB  N*128
  int*      idx = (int*)     (ws + (size_t)84*1024*1024);         //  2 MB  N*4
  _Float16* gbuf= (_Float16*)(ws + (size_t)86*1024*1024);         //  .25MB B*128
  _Float16* wq  = (_Float16*)(ws + (size_t)87*1024*1024);         //  .27MB packed W
  if (ws_size < (size_t)88*1024*1024) return;

  c1.q0 = wq + O10; c1.q1 = wq + O11; c1.q2 = wq + O12;
  c2.q0 = wq + O20; c2.q1 = wq + O21; c2.q2 = wq + O22;
  c3.q0 = wq + O30; c3.q1 = wq + O31; c3.q2 = wq + O32;
  hp.q1 = wq + OH1; hp.q2 = wq + OH2;

  // one-time weight pre-pack into WMMA fragment order
  pack_w_kernel<32, 64 ><<<(32*64  +255)/256, 256, 0, stream>>>((const float*)d_in[3],  wq + O10);
  pack_w_kernel<64, 64 ><<<(64*64  +255)/256, 256, 0, stream>>>((const float*)d_in[9],  wq + O11);
  pack_w_kernel<64, 64 ><<<(64*64  +255)/256, 256, 0, stream>>>((const float*)d_in[15], wq + O12);
  pack_w_kernel<128,128><<<(128*128+255)/256, 256, 0, stream>>>((const float*)d_in[27], wq + O20);
  pack_w_kernel<128,128><<<(128*128+255)/256, 256, 0, stream>>>((const float*)d_in[33], wq + O21);
  pack_w_kernel<128,128><<<(128*128+255)/256, 256, 0, stream>>>((const float*)d_in[39], wq + O22);
  pack_w_kernel<256,128><<<(256*128+255)/256, 256, 0, stream>>>((const float*)d_in[51], wq + O30);
  pack_w_kernel<128,128><<<(128*128+255)/256, 256, 0, stream>>>((const float*)d_in[57], wq + O31);
  pack_w_kernel<128,128><<<(128*128+255)/256, 256, 0, stream>>>((const float*)d_in[63], wq + O32);
  pack_w_kernel<128,64 ><<<(128*64 +255)/256, 256, 0, stream>>>((const float*)d_in[86], wq + OH1);
  pack_w_kernel<64, 64 ><<<(64*64  +255)/256, 256, 0, stream>>>((const float*)d_in[88], wq + OH2);

  gn_kernel<<<BN_, 128, 0, stream>>>(x, gn_b, gn_g, gn_ms, x16);
  edge_mlp_kernel<16, 64 ><<<EN_/64, 128, 0, stream>>>(x16, ei, c1, h1);
  knn_kernel<64 ><<<BN_, 128, 0, stream>>>(h1, idx);
  edge_mlp_kernel<64, 128><<<EN_/64, 128, 0, stream>>>(h1, idx, c2, h2);
  knn_kernel<128><<<BN_, 128, 0, stream>>>(h2, idx);
  edge_mlp_kernel<128,128><<<EN_/64, 128, 0, stream>>>(h2, idx, c3, h3);
  pool_kernel<<<BN_, 128, 0, stream>>>(h3, gbuf);
  head_kernel<<<BN_/64, 128, 0, stream>>>(gbuf, hp, (float*)d_out);
}
